// SpatialGraphTransformer_5884105195701
// MI455X (gfx1250) — compile-verified, matrix operand order checked by reading, not run on hardware
//
#include <hip/hip_runtime.h>
#include <hip/hip_bf16.h>

// ---------------- model constants ----------------
#define D    768
#define NH   12
#define HD   64
#define DFF  3072
#define VSZ  32000
#define NC   8
#define BB   4
#define SS   512
#define MROWS (BB*SS)      // 2048
#define STEPS 8

typedef __attribute__((ext_vector_type(16))) __bf16 v16bf;
typedef __attribute__((ext_vector_type(8)))  float  v8f;

union BF16Frag { v16bf v; unsigned int u[8]; };

__device__ __forceinline__ unsigned int rnebias(unsigned int x) {
  return x + 0x7FFFu + ((x >> 16) & 1u);   // round-to-nearest-even bias
}

__device__ __forceinline__ unsigned short f2bf(float f) {
  return (unsigned short)(rnebias(__float_as_uint(f)) >> 16);
}

// pack two RNE-rounded bf16 into one dword: bytes {b.hi16, a.hi16} via v_perm_b32
__device__ __forceinline__ unsigned int pack2bf(float a, float b) {
  unsigned int ra = rnebias(__float_as_uint(a));
  unsigned int rb = rnebias(__float_as_uint(b));
  return __builtin_amdgcn_perm(rb, ra, 0x07060302u);
}

__device__ __forceinline__ v8f wmma_bf16(v16bf a, v16bf b, v8f c) {
  return __builtin_amdgcn_wmma_f32_16x16x32_bf16(false, a, false, b, (short)0, c, false, false);
}

__device__ __forceinline__ float gelu_tanh(float x) {
  float x3 = x * x * x;
  return 0.5f * x * (1.0f + tanhf(0.7978845608028654f * (x + 0.044715f * x3)));
}

// Fragment layout (ISA 7.12.2): lane row/col = lane&15, half = lane>>4 picks
// K-groups; per lane the 16 bf16 are k in [hf*8,+8) and [16+hf*8,+8): two
// contiguous 16-byte runs -> 2x ds_load_b128 when rows are 16B-aligned.
#define LDR 40   // ushort row stride (80 bytes, 16B aligned)

__device__ __forceinline__ v16bf load_frag16(const unsigned short* base, int row, int ld) {
  int lane = threadIdx.x & 31;
  const unsigned short* p = base + (long long)(row + (lane & 15)) * ld + (lane >> 4) * 8;
  BF16Frag f;
  uint4 lo = *(const uint4*)(p);
  uint4 hi = *(const uint4*)(p + 16);
  f.u[0] = lo.x; f.u[1] = lo.y; f.u[2] = lo.z; f.u[3] = lo.w;
  f.u[4] = hi.x; f.u[5] = hi.y; f.u[6] = hi.z; f.u[7] = hi.w;
  return f.v;
}

// ---------------- misc small kernels ----------------
__global__ void sgt_init_kernel(int* curp, float* lbp) {
  curp[0] = 0; lbp[0] = 0.0f;
}

__global__ void sgt_embed_kernel(const int* __restrict__ ids,
                                 const float* __restrict__ emb,
                                 float* __restrict__ x) {
  long long idx = (long long)blockIdx.x * blockDim.x + threadIdx.x;
  if (idx >= (long long)MROWS * D) return;
  long long row = idx / D; int d = (int)(idx % D);
  x[idx] = emb[(long long)ids[row] * D + d] * 27.712812921102035f;  // sqrt(768)
}

__global__ __launch_bounds__(256)
void sgt_rmsnorm_kernel(const float* __restrict__ x, const float* __restrict__ wbase,
                        const int* __restrict__ curp, int cellStride,
                        float* __restrict__ out) {
  __shared__ float red[8];
  long long row = blockIdx.x;
  const float* xr = x + row * D;
  const float* w = wbase + (curp ? (long long)curp[0] * cellStride : 0);
  float ss = 0.0f;
  for (int i = threadIdx.x; i < D; i += 256) { float v = xr[i]; ss += v * v; }
#pragma unroll
  for (int off = 16; off > 0; off >>= 1) ss += __shfl_xor(ss, off, 32);
  if ((threadIdx.x & 31) == 0) red[threadIdx.x >> 5] = ss;
  __syncthreads();
  float tot = 0.0f;
#pragma unroll
  for (int i = 0; i < 8; ++i) tot += red[i];
  float inv = rsqrtf(tot / (float)D + 1e-6f);
  for (int i = threadIdx.x; i < D; i += 256) out[row * D + i] = xr[i] * inv * w[i];
}

__global__ void sgt_rope_kernel(float* __restrict__ q, float* __restrict__ k,
                                const float* __restrict__ cosT, const float* __restrict__ sinT) {
  long long idx = (long long)blockIdx.x * blockDim.x + threadIdx.x;  // B*S*H*(HD/2)
  if (idx >= (long long)MROWS * NH * (HD / 2)) return;
  int i = (int)(idx & 31);
  int h = (int)((idx >> 5) % NH);
  long long bs = (idx >> 5) / NH;
  int s = (int)(bs % SS);
  float c = cosT[s * 32 + i], sn = sinT[s * 32 + i];
  long long base = (bs * NH + h) * HD;
  float q1 = q[base + i], q2 = q[base + 32 + i];
  q[base + i]      = q1 * c - q2 * sn;
  q[base + 32 + i] = q1 * sn + q2 * c;
  float k1 = k[base + i], k2 = k[base + 32 + i];
  k[base + i]      = k1 * c - k2 * sn;
  k[base + 32 + i] = k1 * sn + k2 * c;
}

// ---------------- generic WMMA GEMM ----------------
// C[M,N] = epi(A[M,K] @ W[K,N] [+ Res]) ; W selected by curp[0]*cellStride
// Block tile 128x128, K-chunks of 32; 8 waves in 4(m) x 2(n), wave tile 32x64.
#define TM 128
#define TN 128
#define TK 32

__global__ __launch_bounds__(256)
void sgt_gemm_kernel(const float* __restrict__ A, const float* __restrict__ Wbase,
                     const int* __restrict__ curp, long long cellStride,
                     float* __restrict__ Cout, const float* __restrict__ Res,
                     int M, int N, int K, int epi) {
  __shared__ __align__(16) unsigned short As[TM * LDR];   // [m][k] 10240 B
  __shared__ __align__(16) unsigned short Bs[TN * LDR];   // [n][k] (transposed) 10240 B
  const float* W = Wbase + (curp ? (long long)curp[0] * cellStride : 0ll);
  int bm = blockIdx.y * TM, bn = blockIdx.x * TN;
  int tid = threadIdx.x, wave = tid >> 5;
  int wm = wave & 3, wn = wave >> 2;         // wave tile: [wm*32, +32) x [wn*64, +64)
  v8f acc[2][4] = {};

  // staging ownership (fixed per thread, pointers advance per K-chunk)
  int am  = tid >> 1;                        // A row 0..127
  int ak0 = (tid & 1) * 16;                  // A k half
  const float* aptr = A + (long long)(bm + am) * K + ak0;
  unsigned short* asdst = As + am * LDR + ak0;

  int wn0 = tid & 127;                       // W column 0..127
  int kp0 = tid >> 7;                        // 0/1: which k-pair set
  const float* wptr = W + (long long)(2 * kp0) * N + bn + wn0;
  unsigned short* bsrow = Bs + wn0 * LDR;

  for (int k0 = 0; k0 < K; k0 += TK) {
    __syncthreads();
    // ---- stage A (128x32): 4x b128 loads, 2x b128 LDS stores per thread
    float4 a0 = *(const float4*)(aptr);
    float4 a1 = *(const float4*)(aptr + 4);
    float4 a2 = *(const float4*)(aptr + 8);
    float4 a3 = *(const float4*)(aptr + 12);
    uint4 p0, p1;
    p0.x = pack2bf(a0.x, a0.y); p0.y = pack2bf(a0.z, a0.w);
    p0.z = pack2bf(a1.x, a1.y); p0.w = pack2bf(a1.z, a1.w);
    p1.x = pack2bf(a2.x, a2.y); p1.y = pack2bf(a2.z, a2.w);
    p1.z = pack2bf(a3.x, a3.y); p1.w = pack2bf(a3.z, a3.w);
    *(uint4*)(asdst)     = p0;
    *(uint4*)(asdst + 8) = p1;
    // ---- stage W (32x128) transposed into Bs[n][k], (k,k+1) packed per dword
#pragma unroll
    for (int j = 0; j < 8; ++j) {
      int k = 2 * (kp0 + 2 * j);             // 0..30 even, split across kp0
      float wlo = wptr[(long long)(4 * j) * N];
      float whi = wptr[(long long)(4 * j) * N + N];
      *(unsigned int*)(bsrow + k) = pack2bf(wlo, whi);
    }
    if (k0 + TK < K)
      __builtin_prefetch(wptr + (long long)TK * N, 0, 1);
    __syncthreads();
    // ---- compute: ds_load_b128 pairs -> 8 WMMAs
    v16bf af0 = load_frag16(As, wm * 32,      LDR);
    v16bf af1 = load_frag16(As, wm * 32 + 16, LDR);
    v16bf bfr[4];
#pragma unroll
    for (int j = 0; j < 4; ++j) bfr[j] = load_frag16(Bs, wn * 64 + j * 16, LDR);
#pragma unroll
    for (int j = 0; j < 4; ++j) {
      acc[0][j] = wmma_bf16(af0, bfr[j], acc[0][j]);
      acc[1][j] = wmma_bf16(af1, bfr[j], acc[1][j]);
    }
    aptr += TK;
    wptr += (long long)TK * N;
  }
  int lane = tid & 31, nn = lane & 15, hf = lane >> 4;
#pragma unroll
  for (int ti = 0; ti < 2; ++ti)
#pragma unroll
    for (int tj = 0; tj < 4; ++tj)
#pragma unroll
      for (int r = 0; r < 8; ++r) {
        int m = bm + wm * 32 + ti * 16 + r + hf * 8;
        int n = bn + wn * 64 + tj * 16 + nn;
        float v = acc[ti][tj][r];
        if (epi == 1) v = gelu_tanh(v);
        else if (epi == 2) v += Res[(long long)m * N + n];
        Cout[(long long)m * N + n] = v;
      }
}

// ---------------- flash attention (1 wave per (b,h,16 queries)) ----------------
__global__ __launch_bounds__(32)
void sgt_attn_kernel(const float* __restrict__ q, const float* __restrict__ k,
                     const float* __restrict__ v, float* __restrict__ o) {
  __shared__ __align__(16) unsigned short Plds[16 * LDR];
  int qt = blockIdx.x % (SS / 16);
  int h  = (blockIdx.x / (SS / 16)) % NH;
  int b  = blockIdx.x / ((SS / 16) * NH);
  int lane = threadIdx.x;
  int nn = lane & 15, hf = lane >> 4;

  // Q fragments (A layout): per-lane runs are 8 consecutive d -> float4 loads
  v16bf aq[2];
  {
    const float* qrow = q + (((long long)(b * SS + qt * 16 + nn) * NH + h) * HD);
#pragma unroll
    for (int c = 0; c < 2; ++c) {
      const float* pA = qrow + c * 32 + hf * 8;
      const float* pB = qrow + c * 32 + 16 + hf * 8;
      float4 x0 = *(const float4*)(pA), x1 = *(const float4*)(pA + 4);
      float4 x2 = *(const float4*)(pB), x3 = *(const float4*)(pB + 4);
      BF16Frag t;
      t.u[0] = pack2bf(x0.x, x0.y); t.u[1] = pack2bf(x0.z, x0.w);
      t.u[2] = pack2bf(x1.x, x1.y); t.u[3] = pack2bf(x1.z, x1.w);
      t.u[4] = pack2bf(x2.x, x2.y); t.u[5] = pack2bf(x2.z, x2.w);
      t.u[6] = pack2bf(x3.x, x3.y); t.u[7] = pack2bf(x3.z, x3.w);
      aq[c] = t.v;
    }
  }

  v8f oacc[4] = {};
  float rowmax[8], rowsum[8];
#pragma unroll
  for (int r = 0; r < 8; ++r) { rowmax[r] = -3.0e30f; rowsum[r] = 0.0f; }

  int qend = qt * 16 + 15;
  int nkb = qend / 32 + 1;
  const float scale = 0.125f;  // 1/sqrt(64)

  for (int kb = 0; kb < nkb; ++kb) {
    int kbase = kb * 32;
    // ---- scores: S = Q K^T (16x32 block): 2 key subtiles x 2 d-chunks
    v8f sc[2] = {};
#pragma unroll
    for (int sub = 0; sub < 2; ++sub) {
      const float* krow = k + (((long long)(b * SS + kbase + sub * 16 + nn) * NH + h) * HD);
#pragma unroll
      for (int c = 0; c < 2; ++c) {
        const float* pA = krow + c * 32 + hf * 8;
        const float* pB = krow + c * 32 + 16 + hf * 8;
        float4 x0 = *(const float4*)(pA), x1 = *(const float4*)(pA + 4);
        float4 x2 = *(const float4*)(pB), x3 = *(const float4*)(pB + 4);
        BF16Frag t;
        t.u[0] = pack2bf(x0.x, x0.y); t.u[1] = pack2bf(x0.z, x0.w);
        t.u[2] = pack2bf(x1.x, x1.y); t.u[3] = pack2bf(x1.z, x1.w);
        t.u[4] = pack2bf(x2.x, x2.y); t.u[5] = pack2bf(x2.z, x2.w);
        t.u[6] = pack2bf(x3.x, x3.y); t.u[7] = pack2bf(x3.z, x3.w);
        sc[sub] = wmma_bf16(aq[c], t.v, sc[sub]);
      }
    }
    // ---- online softmax (rows live across 16 lanes of each half-wave)
#pragma unroll
    for (int r = 0; r < 8; ++r) {
      int m = qt * 16 + r + hf * 8;
      float best = rowmax[r];
#pragma unroll
      for (int sub = 0; sub < 2; ++sub) {
        int kk = kbase + sub * 16 + nn;
        float val = sc[sub][r] * scale;
        if (kk > m) val = -1.0e9f;
        sc[sub][r] = val;
        best = fmaxf(best, val);
      }
#pragma unroll
      for (int off = 1; off < 16; off <<= 1) best = fmaxf(best, __shfl_xor(best, off, 32));
      float corr = __expf(rowmax[r] - best);
      rowsum[r] *= corr;
      rowmax[r] = best;
#pragma unroll
      for (int nt = 0; nt < 4; ++nt) oacc[nt][r] *= corr;
      float lsum = 0.0f;
#pragma unroll
      for (int sub = 0; sub < 2; ++sub) {
        float p = __expf(sc[sub][r] - best);
        sc[sub][r] = p;
        lsum += p;
      }
#pragma unroll
      for (int off = 1; off < 16; off <<= 1) lsum += __shfl_xor(lsum, off, 32);
      rowsum[r] += lsum;
    }
    // ---- re-layout P (C layout -> A layout) through LDS
#pragma unroll
    for (int sub = 0; sub < 2; ++sub)
#pragma unroll
      for (int r = 0; r < 8; ++r)
        Plds[(r + hf * 8) * LDR + sub * 16 + nn] = f2bf(sc[sub][r]);
    __syncthreads();
    v16bf pf;
    {
      const unsigned short* p = Plds + nn * LDR + hf * 8;
      BF16Frag t;
      uint4 lo = *(const uint4*)(p);
      uint4 hi = *(const uint4*)(p + 16);
      t.u[0] = lo.x; t.u[1] = lo.y; t.u[2] = lo.z; t.u[3] = lo.w;
      t.u[4] = hi.x; t.u[5] = hi.y; t.u[6] = hi.z; t.u[7] = hi.w;
      pf = t.v;
    }
    // ---- O += P @ V  (4 column tiles of HD); V pairs stride across keys
#pragma unroll
    for (int nt = 0; nt < 4; ++nt) {
      BF16Frag t;
#pragma unroll
      for (int d = 0; d < 8; ++d) {
        int kk = kbase + ((d >> 2) << 4) + hf * 8 + ((d & 3) << 1);
        const float* v0 = v + (((long long)(b * SS + kk) * NH + h) * HD) + nt * 16 + nn;
        const float* v1 = v0 + (long long)NH * HD;  // next key
        t.u[d] = pack2bf(*v0, *v1);
      }
      oacc[nt] = wmma_bf16(pf, t.v, oacc[nt]);
    }
    __syncthreads();
  }
  // ---- write O
#pragma unroll
  for (int nt = 0; nt < 4; ++nt)
#pragma unroll
    for (int r = 0; r < 8; ++r) {
      int m = qt * 16 + r + hf * 8;
      int d = nt * 16 + nn;
      o[((long long)(b * SS + m) * NH + h) * HD + d] = oacc[nt][r] / rowsum[r];
    }
}

// ---------------- routing / gating ----------------
__global__ void sgt_rowmean_kernel(const float* __restrict__ x, float* __restrict__ xmean) {
  int idx = blockIdx.x * blockDim.x + threadIdx.x;  // B*D
  if (idx >= BB * D) return;
  int b = idx / D, d = idx % D;
  float s = 0.0f;
  for (int t = 0; t < SS; ++t) s += x[((long long)b * SS + t) * D + d];
  xmean[idx] = s / (float)SS;
}

__global__ __launch_bounds__(64)
void sgt_gate_kernel(const float* __restrict__ xmean, const float* __restrict__ gate_w,
                     const int* __restrict__ neighbors, int* curp, float* lbp) {
  __shared__ float scores[BB][NC + 1];
  int t = threadIdx.x;
  if (t < BB * (NC + 1)) {
    int b = t / (NC + 1), c = t % (NC + 1);
    float s = 0.0f;
    for (int d = 0; d < D; ++d) s += xmean[b * D + d] * gate_w[d * (NC + 1) + c];
    scores[b][c] = s;
  }
  __syncthreads();
  if (t == 0) {
    float avg[NC + 1];
#pragma unroll
    for (int c = 0; c <= NC; ++c) avg[c] = 0.0f;
    for (int b = 0; b < BB; ++b) {
      float mx = -3.0e30f;
      for (int c = 0; c <= NC; ++c) mx = fmaxf(mx, scores[b][c]);
      float e[NC + 1], sum = 0.0f;
      for (int c = 0; c <= NC; ++c) { e[c] = __expf(scores[b][c] - mx); sum += e[c]; }
      for (int c = 0; c <= NC; ++c) avg[c] += e[c] / sum;
    }
    float qd = 0.0f;
    for (int c = 0; c <= NC; ++c) { avg[c] /= (float)BB; qd += avg[c] * avg[c]; }
    lbp[0] += (float)(NC + 1) * qd;
    int cur = curp[0];
    float bests = -3.0e30f; int bestn = 0;
    for (int j = 0; j < 4; ++j) {
      int ngj = neighbors[cur * 4 + j];
      float ns = 0.0f;
      for (int b = 0; b < BB; ++b) ns += scores[b][ngj];
      ns /= (float)BB;
      if (ns > bests) { bests = ns; bestn = ngj; }
    }
    curp[0] = bestn;
  }
}

__global__ void sgt_fin_kernel(const float* lbp, float* outp) {
  outp[0] = lbp[0] / (float)STEPS;
}

// ---------------- host launcher ----------------
extern "C" void kernel_launch(void* const* d_in, const int* in_sizes, int n_in,
                              void* d_out, int out_size, void* d_ws, size_t ws_size,
                              hipStream_t stream) {
  const int*   ids         = (const int*)d_in[0];
  const float* tok_emb     = (const float*)d_in[1];
  const float* Wq          = (const float*)d_in[2];
  const float* Wk          = (const float*)d_in[3];
  const float* Wv          = (const float*)d_in[4];
  const float* Wo          = (const float*)d_in[5];
  const float* attn_norm_w = (const float*)d_in[6];
  const float* ffn_norm_w  = (const float*)d_in[7];
  const float* W1          = (const float*)d_in[8];
  const float* W2          = (const float*)d_in[9];
  const float* gate_w      = (const float*)d_in[10];
  const float* final_norm_w= (const float*)d_in[11];
  const float* fc_out_w    = (const float*)d_in[12];
  const float* cosT        = (const float*)d_in[13];
  const float* sinT        = (const float*)d_in[14];
  const int*   neigh       = (const int*)d_in[15];
  float* out = (float*)d_out;
  float* wsf = (float*)d_ws;

  const long long BSD = (long long)MROWS * D;     // 1,572,864
  float* x   = wsf;
  float* hbf = x   + BSD;
  float* qb  = hbf + BSD;
  float* kb  = qb  + BSD;
  float* vb  = kb  + BSD;
  float* ob  = vb  + BSD;
  float* gb  = ob  + BSD;                          // B*S*DFF
  float* xmean = gb + (long long)MROWS * DFF;
  int*   curp  = (int*)(xmean + BB * D);
  float* lbp   = (float*)(curp + 1);

  sgt_init_kernel<<<1, 1, 0, stream>>>(curp, lbp);
  {
    long long n = BSD;
    sgt_embed_kernel<<<(unsigned)((n + 255) / 256), 256, 0, stream>>>(ids, tok_emb, x);
  }

  dim3 gD(D / TN, MROWS / TM);       // 6 x 16
  dim3 gF(DFF / TN, MROWS / TM);     // 24 x 16
  dim3 gV(VSZ / TN, MROWS / TM);     // 250 x 16

  for (int step = 0; step < STEPS; ++step) {
    sgt_rmsnorm_kernel<<<MROWS, 256, 0, stream>>>(x, attn_norm_w, curp, D, hbf);
    sgt_gemm_kernel<<<gD, 256, 0, stream>>>(hbf, Wq, curp, (long long)D * D, qb, nullptr, MROWS, D, D, 0);
    sgt_gemm_kernel<<<gD, 256, 0, stream>>>(hbf, Wk, curp, (long long)D * D, kb, nullptr, MROWS, D, D, 0);
    sgt_gemm_kernel<<<gD, 256, 0, stream>>>(hbf, Wv, curp, (long long)D * D, vb, nullptr, MROWS, D, D, 0);
    {
      long long n = (long long)MROWS * NH * (HD / 2);
      sgt_rope_kernel<<<(unsigned)((n + 255) / 256), 256, 0, stream>>>(qb, kb, cosT, sinT);
    }
    sgt_attn_kernel<<<BB * NH * (SS / 16), 32, 0, stream>>>(qb, kb, vb, ob);
    sgt_gemm_kernel<<<gD, 256, 0, stream>>>(ob, Wo, curp, (long long)D * D, x, x, MROWS, D, D, 2);
    sgt_rmsnorm_kernel<<<MROWS, 256, 0, stream>>>(x, ffn_norm_w, curp, D, hbf);
    sgt_gemm_kernel<<<gF, 256, 0, stream>>>(hbf, W1, curp, (long long)D * DFF, gb, nullptr, MROWS, DFF, D, 1);
    sgt_gemm_kernel<<<gD, 256, 0, stream>>>(gb, W2, curp, (long long)DFF * D, x, x, MROWS, D, DFF, 2);
    sgt_rowmean_kernel<<<(BB * D + 255) / 256, 256, 0, stream>>>(x, xmean);
    sgt_gate_kernel<<<1, 64, 0, stream>>>(xmean, gate_w, neigh, curp, lbp);
  }

  sgt_rmsnorm_kernel<<<MROWS, 256, 0, stream>>>(x, final_norm_w, nullptr, 0, hbf);
  sgt_gemm_kernel<<<gV, 256, 0, stream>>>(hbf, fc_out_w, nullptr, 0, out, nullptr, MROWS, VSZ, D, 0);
  sgt_fin_kernel<<<1, 1, 0, stream>>>(lbp, out + (long long)MROWS * VSZ);
}
